// NMT_51496657879539
// MI455X (gfx1250) — compile-verified, matrix operand order checked by reading, not run on hardware
//
#include <hip/hip_runtime.h>
#include <hip/hip_bf16.h>

// ---------------------------------------------------------------------------
// NMT seq2seq for gfx1250 (MI455X). All GEMMs via v_wmma_f32_16x16x32_bf16.
// Weights converted fp32->bf16 once (W_vocab: 64MB bf16 -> L2-resident).
// Big GEMMs: 2x2 tiles/wave (halved L2 traffic). Sequential step GEMMs:
// 1 tile/wave (max parallelism). Vocab GEMM fused with online log-softmax.
// ---------------------------------------------------------------------------

typedef __hip_bfloat16 bf16;
typedef __attribute__((ext_vector_type(16))) __bf16 v16bf;
typedef __attribute__((ext_vector_type(8)))  float  v8f;

union BF16x16 { uint4 q[2]; v16bf v; };

// CDNA5 16-bit A/B fragment (16x32, MxK / NxK with W row-major):
// lane l: r = l&15 (row of A / row of W), hi = l>>4;
// halves 0..7  = k + hi*8 + 0..7, halves 8..15 = k + hi*8 + 16..23
__device__ __forceinline__ v16bf frag_load(const bf16* base, int ld, int lane) {
    int r = lane & 15, hi = lane >> 4;
    const bf16* p = base + (size_t)r * ld + hi * 8;
    BF16x16 f;
    f.q[0] = *reinterpret_cast<const uint4*>(p);
    f.q[1] = *reinterpret_cast<const uint4*>(p + 16);
    return f.v;
}

__device__ __forceinline__ float sigf(float x) { return 1.f / (1.f + expf(-x)); }

// Epilogue for one 16x16 f32 accumulator tile.
__device__ __forceinline__ void gemm_store(
    v8f acc, int m0, int n0, int lane,
    float* C, int ldc, bf16* Cbf, int ldcb,
    const float* bias, const float* Cadd, int ldadd, int caddMod, int act) {
    int r = lane & 15, hi = lane >> 4;
    int col = n0 + r;
    float bv = bias ? bias[col] : 0.f;
#pragma unroll
    for (int v = 0; v < 8; ++v) {
        int row = m0 + v + hi * 8;
        float x = acc[v] + bv;
        if (Cadd) {
            int cr = caddMod ? (row % caddMod) : row;
            x += Cadd[(size_t)cr * ldadd + col];
        }
        if (act == 1) x = tanhf(x);
        if (C)   C[(size_t)row * ldc + col] = x;
        if (Cbf) Cbf[(size_t)row * ldcb + col] = __float2bfloat16(x);
    }
}

// ---------------------------------------------------------------------------
// 1 tile / wave GEMM (max parallelism; used for the sequential M=64 steps):
// C[M,N] = act( A[M,K](bf16) @ W[N,K]^T(bf16) + bias + Cadd )
// grid = (N/16, M/16), block = 32 (one wave).
// ---------------------------------------------------------------------------
__global__ __launch_bounds__(32) void gemm_wmma(
    const bf16* __restrict__ Amat, int lda,
    const bf16* __restrict__ Wmat, int ldw,
    float* __restrict__ C, int ldc,
    bf16* __restrict__ Cbf, int ldcb,
    const float* __restrict__ bias,
    const float* __restrict__ Cadd, int ldadd, int caddMod,
    int K, int act)
{
    int lane = threadIdx.x;
    int n0 = blockIdx.x * 16;
    int m0 = blockIdx.y * 16;
    v8f acc = {0.f,0.f,0.f,0.f,0.f,0.f,0.f,0.f};
    const bf16* arow = Amat + (size_t)m0 * lda;
    const bf16* wrow = Wmat + (size_t)n0 * ldw;
    for (int kb = 0; kb < K; kb += 32) {
        v16bf af = frag_load(arow + kb, lda, lane);
        v16bf bfr = frag_load(wrow + kb, ldw, lane);
        acc = __builtin_amdgcn_wmma_f32_16x16x32_bf16(false, af, false, bfr,
                                                      (short)0, acc, false, false);
    }
    gemm_store(acc, m0, n0, lane, C, ldc, Cbf, ldcb, bias, Cadd, ldadd, caddMod, act);
}

// ---------------------------------------------------------------------------
// 2x2 tiles / wave GEMM (32x32 output per wave; each fragment reused twice ->
// 8 b128 loads per 4 WMMAs instead of 16). Used for the large parallel GEMMs.
// grid = (N/32, M/32), block = 32. M,N multiples of 32.
// ---------------------------------------------------------------------------
__global__ __launch_bounds__(32) void gemm_wmma_2x2(
    const bf16* __restrict__ Amat, int lda,
    const bf16* __restrict__ Wmat, int ldw,
    float* __restrict__ C, int ldc,
    bf16* __restrict__ Cbf, int ldcb,
    const float* __restrict__ bias,
    const float* __restrict__ Cadd, int ldadd, int caddMod,
    int K, int act)
{
    int lane = threadIdx.x;
    int n0 = blockIdx.x * 32;
    int m0 = blockIdx.y * 32;
    v8f a00 = {0.f,0.f,0.f,0.f,0.f,0.f,0.f,0.f};
    v8f a01 = a00, a10 = a00, a11 = a00;
    const bf16* arow0 = Amat + (size_t)m0 * lda;
    const bf16* arow1 = Amat + (size_t)(m0 + 16) * lda;
    const bf16* wrow0 = Wmat + (size_t)n0 * ldw;
    const bf16* wrow1 = Wmat + (size_t)(n0 + 16) * ldw;
    for (int kb = 0; kb < K; kb += 32) {
        v16bf af0 = frag_load(arow0 + kb, lda, lane);
        v16bf af1 = frag_load(arow1 + kb, lda, lane);
        v16bf bf0 = frag_load(wrow0 + kb, ldw, lane);
        v16bf bf1 = frag_load(wrow1 + kb, ldw, lane);
        a00 = __builtin_amdgcn_wmma_f32_16x16x32_bf16(false, af0, false, bf0,
                                                      (short)0, a00, false, false);
        a01 = __builtin_amdgcn_wmma_f32_16x16x32_bf16(false, af0, false, bf1,
                                                      (short)0, a01, false, false);
        a10 = __builtin_amdgcn_wmma_f32_16x16x32_bf16(false, af1, false, bf0,
                                                      (short)0, a10, false, false);
        a11 = __builtin_amdgcn_wmma_f32_16x16x32_bf16(false, af1, false, bf1,
                                                      (short)0, a11, false, false);
    }
    gemm_store(a00, m0,      n0,      lane, C, ldc, Cbf, ldcb, bias, Cadd, ldadd, caddMod, act);
    gemm_store(a01, m0,      n0 + 16, lane, C, ldc, Cbf, ldcb, bias, Cadd, ldadd, caddMod, act);
    gemm_store(a10, m0 + 16, n0,      lane, C, ldc, Cbf, ldcb, bias, Cadd, ldadd, caddMod, act);
    gemm_store(a11, m0 + 16, n0 + 16, lane, C, ldc, Cbf, ldcb, bias, Cadd, ldadd, caddMod, act);
}

// ---------------------------------------------------------------------------
// Elementwise helpers
// ---------------------------------------------------------------------------
__global__ void cvt_bf16_k(const float* __restrict__ s, bf16* __restrict__ d, long n) {
    long i = (long)blockIdx.x * blockDim.x + threadIdx.x;
    long st = (long)gridDim.x * blockDim.x;
    for (; i < n; i += st) d[i] = __float2bfloat16(s[i]);
}
__global__ void zero_f32_k(float* p, long n) {
    long i = (long)blockIdx.x * blockDim.x + threadIdx.x;
    long st = (long)gridDim.x * blockDim.x;
    for (; i < n; i += st) p[i] = 0.f;
}
__global__ void zero_bf16_k(bf16* p, long n) {
    long i = (long)blockIdx.x * blockDim.x + threadIdx.x;
    long st = (long)gridDim.x * blockDim.x;
    for (; i < n; i += st) p[i] = __float2bfloat16(0.f);
}

// Source embeddings: X[t,b,:] and Xrev[t,b,:] = X[rev_idx[t,b],b,:]
__global__ void build_src_k(const int* __restrict__ src, const int* __restrict__ lens,
                            const float* __restrict__ emb,
                            bf16* __restrict__ Xf, bf16* __restrict__ Xr) {
    long n = (long)48 * 64 * 512;
    long i = (long)blockIdx.x * blockDim.x + threadIdx.x;
    long st = (long)gridDim.x * blockDim.x;
    for (; i < n; i += st) {
        int e = (int)(i & 511);
        long bt = i >> 9;
        int b = (int)(bt % 64), t = (int)(bt / 64);
        int tok = src[t * 64 + b];
        Xf[i] = __float2bfloat16(emb[(size_t)tok * 512 + e]);
        int len = lens[b];
        int rt = (t < len) ? (len - 1 - t) : t;
        int tok2 = src[rt * 64 + b];
        Xr[i] = __float2bfloat16(emb[(size_t)tok2 * 512 + e]);
    }
}

__global__ void build_tgt_k(const int* __restrict__ tgt, const float* __restrict__ emb,
                            bf16* __restrict__ Y) {
    long n = (long)47 * 64 * 512;
    long i = (long)blockIdx.x * blockDim.x + threadIdx.x;
    long st = (long)gridDim.x * blockDim.x;
    for (; i < n; i += st) {
        int e = (int)(i & 511);
        long bt = i >> 9;
        int b = (int)(bt % 64), t = (int)(bt / 64);
        int tok = tgt[t * 64 + b];
        Y[i] = __float2bfloat16(emb[(size_t)tok * 512 + e]);
    }
}

// Encoder LSTM pointwise (masked carry). Writes h/c (f32), bf16 h/c into
// strided concat buffers, and masked output into enc_hiddens[b, pos, dir*HE + j].
__global__ void enc_pointwise_k(const float* __restrict__ gates,
                                float* __restrict__ h, float* __restrict__ c,
                                bf16* __restrict__ hcat, bf16* __restrict__ ccat,
                                bf16* __restrict__ enc_hid,
                                const int* __restrict__ lens, int t, int dir) {
    int idx = blockIdx.x * blockDim.x + threadIdx.x;
    if (idx >= 64 * 1024) return;
    int b = idx >> 10, j = idx & 1023;
    const float* g = gates + (size_t)b * 4096;
    float gi = g[j], gf = g[1024 + j], gg = g[2048 + j], go = g[3072 + j];
    float c2 = sigf(gf) * c[idx] + sigf(gi) * tanhf(gg);
    float h2 = sigf(go) * tanhf(c2);
    int len = lens[b];
    bool m = t < len;
    float hn = m ? h2 : h[idx];
    float cn = m ? c2 : c[idx];
    h[idx] = hn; c[idx] = cn;
    hcat[(size_t)b * 2048 + j] = __float2bfloat16(hn);
    ccat[(size_t)b * 2048 + j] = __float2bfloat16(cn);
    int p = dir ? (m ? (len - 1 - t) : t) : t;
    enc_hid[((size_t)b * 48 + p) * 2048 + (size_t)dir * 1024 + j] =
        __float2bfloat16(m ? h2 : 0.f);
}

// Decoder LSTM pointwise (constant c0; no mask). Writes h_t f32 and bf16 h_t
// into Abuf columns [2048, 3072).
__global__ void dec_pointwise_k(const float* __restrict__ gates,
                                const float* __restrict__ c0,
                                float* __restrict__ ht, bf16* __restrict__ Abuf) {
    int idx = blockIdx.x * blockDim.x + threadIdx.x;
    if (idx >= 64 * 1024) return;
    int b = idx >> 10, j = idx & 1023;
    const float* g = gates + (size_t)b * 4096;
    float gi = g[j], gf = g[1024 + j], gg = g[2048 + j], go = g[3072 + j];
    float c2 = sigf(gf) * c0[idx] + sigf(gi) * tanhf(gg);
    float h2 = sigf(go) * tanhf(c2);
    ht[idx] = h2;
    Abuf[(size_t)b * 3072 + 2048 + j] = __float2bfloat16(h2);
}

// Attention: one block per batch element. e = enc_proj . h_t, masked softmax,
// a = sum_s alpha * enc_hiddens -> Abuf columns [0,2048).
__global__ void attention_k(const float* __restrict__ ht,
                            const float* __restrict__ enc_proj,
                            const bf16* __restrict__ enc_hid,
                            const int* __restrict__ lens,
                            bf16* __restrict__ Abuf) {
    __shared__ float hs[1024];
    __shared__ float ev[48];
    int b = blockIdx.x;
    int tid = threadIdx.x;
    for (int i = tid; i < 1024; i += 256) hs[i] = ht[(size_t)b * 1024 + i];
    __syncthreads();
    int wave = tid >> 5, lane = tid & 31;
    int len = lens[b];
    for (int s = wave; s < 48; s += 8) {
        const float* ep = enc_proj + ((size_t)b * 48 + s) * 1024;
        float acc = 0.f;
        for (int i = lane; i < 1024; i += 32) acc += ep[i] * hs[i];
        for (int msk = 16; msk; msk >>= 1) acc += __shfl_xor(acc, msk, 32);
        if (lane == 0) ev[s] = (s < len) ? acc : -__builtin_inff();
    }
    __syncthreads();
    if (tid == 0) {
        float mx = -__builtin_inff();
        for (int s = 0; s < 48; ++s) mx = fmaxf(mx, ev[s]);
        float sum = 0.f;
        for (int s = 0; s < 48; ++s) { float p = expf(ev[s] - mx); ev[s] = p; sum += p; }
        float inv = 1.f / sum;
        for (int s = 0; s < 48; ++s) ev[s] *= inv;
    }
    __syncthreads();
    for (int d = tid; d < 2048; d += 256) {
        float a = 0.f;
        const bf16* eh = enc_hid + (size_t)b * 48 * 2048 + d;
        for (int s = 0; s < 48; ++s) a += ev[s] * __bfloat162float(eh[(size_t)s * 2048]);
        Abuf[(size_t)b * 3072 + d] = __float2bfloat16(a);
    }
}

// ---------------------------------------------------------------------------
// Fused vocab GEMM + online log-softmax + gold gather.
// Block = 16 rows of `combined` (A tile cached in LDS), 8 waves stream the
// 32000-wide projection; each wave handles TWO 16-col subtiles per A-fragment
// read (1 ds_read pair + 2 global pairs + 2 WMMAs per k-step), tracking
// running max / sum-exp per row and capturing the gold logit.
// grid = 3008/16 = 188 blocks.
// ---------------------------------------------------------------------------
__global__ __launch_bounds__(256) void vocab_loss_k(
    const bf16* __restrict__ combined, const bf16* __restrict__ Wv,
    const int* __restrict__ target, float* __restrict__ out) {
    __shared__ __align__(16) bf16 aT[16 * 1024];
    __shared__ int goldcol[16];
    __shared__ float goldval[16];
    __shared__ float smax[8 * 16];
    __shared__ float ssum[8 * 16];

    int r0 = blockIdx.x * 16;
    int tid = threadIdx.x;

    const unsigned* srcw = (const unsigned*)(combined + (size_t)r0 * 1024);
    unsigned* dstw = (unsigned*)aT;
    for (int i = tid; i < 16 * 512; i += 256) dstw[i] = srcw[i];
    if (tid < 16) {
        int gm = r0 + tid;
        int t = gm / 64, b = gm % 64;
        goldcol[tid] = target[(t + 1) * 64 + b];
        goldval[tid] = 0.f;
    }
    __syncthreads();

    int wave = tid >> 5, lane = tid & 31;
    int r = lane & 15, hi = lane >> 4;
    float run_m[8], run_s[8];
#pragma unroll
    for (int v = 0; v < 8; ++v) { run_m[v] = -__builtin_inff(); run_s[v] = 0.f; }

    // 32000 = 256 * 125 -> each wave: 125 iterations of two 16-col subtiles.
    for (int nb = wave * 32; nb < 32000; nb += 256) {
        v8f acc0 = {0.f,0.f,0.f,0.f,0.f,0.f,0.f,0.f};
        v8f acc1 = acc0;
        const bf16* wrow0 = Wv + (size_t)nb * 1024;
        const bf16* wrow1 = Wv + (size_t)(nb + 16) * 1024;
        for (int kb = 0; kb < 1024; kb += 32) {
            v16bf af  = frag_load(aT + kb, 1024, lane);
            v16bf bf0 = frag_load(wrow0 + kb, 1024, lane);
            v16bf bf1 = frag_load(wrow1 + kb, 1024, lane);
            acc0 = __builtin_amdgcn_wmma_f32_16x16x32_bf16(false, af, false, bf0,
                                                           (short)0, acc0, false, false);
            acc1 = __builtin_amdgcn_wmma_f32_16x16x32_bf16(false, af, false, bf1,
                                                           (short)0, acc1, false, false);
        }
#pragma unroll
        for (int half = 0; half < 2; ++half) {
            v8f acc = half ? acc1 : acc0;
            int col = nb + half * 16 + r;
#pragma unroll
            for (int v = 0; v < 8; ++v) {
                int row = v + hi * 8;
                if (col == goldcol[row]) goldval[row] = acc[v];  // unique col -> no race
            }
#pragma unroll
            for (int v = 0; v < 8; ++v) {
                float x = acc[v];
                float mx = x;
                for (int msk = 1; msk < 16; msk <<= 1)
                    mx = fmaxf(mx, __shfl_xor(mx, msk, 32));
                float nm = fmaxf(run_m[v], mx);
                float p = expf(x - nm);
                for (int msk = 1; msk < 16; msk <<= 1)
                    p += __shfl_xor(p, msk, 32);
                run_s[v] = run_s[v] * expf(run_m[v] - nm) + p;
                run_m[v] = nm;
            }
        }
    }
    if (r == 0) {
#pragma unroll
        for (int v = 0; v < 8; ++v) {
            smax[wave * 16 + hi * 8 + v] = run_m[v];
            ssum[wave * 16 + hi * 8 + v] = run_s[v];
        }
    }
    __syncthreads();
    if (tid < 16) {
        float M = -__builtin_inff();
        for (int w = 0; w < 8; ++w) M = fmaxf(M, smax[w * 16 + tid]);
        float Ssum = 0.f;
        for (int w = 0; w < 8; ++w) Ssum += ssum[w * 16 + tid] * expf(smax[w * 16 + tid] - M);
        float logZ = M + logf(Ssum);
        int gm = r0 + tid;
        int t = gm / 64, b = gm % 64;
        int tok = target[(t + 1) * 64 + b];
        if (tok != 0) atomicAdd(&out[b], goldval[tid] - logZ);
    }
}

// ---------------------------------------------------------------------------
// Host orchestration
// ---------------------------------------------------------------------------
extern "C" void kernel_launch(void* const* d_in, const int* in_sizes, int n_in,
                              void* d_out, int out_size, void* d_ws, size_t ws_size,
                              hipStream_t stream) {
    (void)in_sizes; (void)n_in; (void)out_size; (void)ws_size;
    const int S = 48, T = 48, B = 64, E = 512, HE = 1024, HD = 1024, V = 32000;
    const int G = 4 * HE;  // 4096

    const int*   src      = (const int*)d_in[0];
    const int*   tgt      = (const int*)d_in[1];
    const int*   lens     = (const int*)d_in[2];
    const float* src_emb  = (const float*)d_in[3];
    const float* tgt_emb  = (const float*)d_in[4];
    const float* eWih_f   = (const float*)d_in[5];
    const float* eWhh_f   = (const float*)d_in[6];
    const float* eb_f     = (const float*)d_in[7];
    const float* eWih_b   = (const float*)d_in[8];
    const float* eWhh_b   = (const float*)d_in[9];
    const float* eb_b     = (const float*)d_in[10];
    const float* dWih     = (const float*)d_in[11];
    const float* dWhh     = (const float*)d_in[12];
    const float* db       = (const float*)d_in[13];
    const float* W_h      = (const float*)d_in[14];
    const float* W_c      = (const float*)d_in[15];
    const float* W_att    = (const float*)d_in[16];
    const float* W_comb   = (const float*)d_in[17];
    const float* W_vocab  = (const float*)d_in[18];

    size_t off = 0;
    auto WS = [&](size_t bytes) -> void* {
        void* p = (char*)d_ws + off;
        off += (bytes + 255) & ~(size_t)255;
        return p;
    };

    bf16* wih_f  = (bf16*)WS((size_t)G * E * 2);
    bf16* whh_f  = (bf16*)WS((size_t)G * HE * 2);
    bf16* wih_b  = (bf16*)WS((size_t)G * E * 2);
    bf16* whh_b  = (bf16*)WS((size_t)G * HE * 2);
    bf16* dwih   = (bf16*)WS((size_t)G * (E + HD) * 2);
    bf16* dwhh   = (bf16*)WS((size_t)G * HD * 2);
    bf16* wh     = (bf16*)WS((size_t)HD * 2 * HE * 2);
    bf16* wc     = (bf16*)WS((size_t)HD * 2 * HE * 2);
    bf16* watt   = (bf16*)WS((size_t)HD * 2 * HE * 2);
    bf16* wcomb  = (bf16*)WS((size_t)HE * (2 * HE + HD) * 2);
    bf16* wvoc   = (bf16*)WS((size_t)V * HD * 2);
    bf16* Xf     = (bf16*)WS((size_t)S * B * E * 2);
    bf16* Xr     = (bf16*)WS((size_t)S * B * E * 2);
    bf16* Ybuf   = (bf16*)WS((size_t)(T - 1) * B * E * 2);
    float* Gx_f  = (float*)WS((size_t)S * B * G * 4);
    float* Gx_b  = (float*)WS((size_t)S * B * G * 4);
    float* Gy    = (float*)WS((size_t)(T - 1) * B * G * 4);
    float* gates = (float*)WS((size_t)B * G * 4);
    float* hbuf  = (float*)WS((size_t)B * HE * 4);
    float* cbuf  = (float*)WS((size_t)B * HE * 4);
    bf16* hcat   = (bf16*)WS((size_t)B * 2 * HE * 2);
    bf16* ccat   = (bf16*)WS((size_t)B * 2 * HE * 2);
    bf16* enc_hid= (bf16*)WS((size_t)B * S * 2 * HE * 2);
    float* encproj=(float*)WS((size_t)B * S * HD * 4);
    bf16* dech0  = (bf16*)WS((size_t)B * HD * 2);
    float* decc0 = (float*)WS((size_t)B * HD * 4);
    float* Ghh0  = (float*)WS((size_t)B * G * 4);
    float* htbuf = (float*)WS((size_t)B * HD * 4);
    bf16* Abuf   = (bf16*)WS((size_t)B * (2 * HE + HD) * 2);
    bf16* comb   = (bf16*)WS((size_t)(T - 1) * B * HD * 2);
    bf16* zerob  = (bf16*)WS((size_t)B * HD * 2);

    auto cvt = [&](const float* s, bf16* d, long n) {
        int g = (int)((n + 1023) / 1024); if (g > 2048) g = 2048;
        cvt_bf16_k<<<g, 256, 0, stream>>>(s, d, n);
    };
    auto gemm1 = [&](const bf16* A, int lda, const bf16* W, int ldw,
                     float* C, int ldc, bf16* Cbf, int ldcb,
                     const float* bias, const float* Cadd, int ldadd, int cmod,
                     int M, int N, int K, int act) {
        gemm_wmma<<<dim3(N / 16, M / 16), dim3(32), 0, stream>>>(
            A, lda, W, ldw, C, ldc, Cbf, ldcb, bias, Cadd, ldadd, cmod, K, act);
    };
    auto gemm2 = [&](const bf16* A, int lda, const bf16* W, int ldw,
                     float* C, int ldc, bf16* Cbf, int ldcb,
                     const float* bias, const float* Cadd, int ldadd, int cmod,
                     int M, int N, int K, int act) {
        gemm_wmma_2x2<<<dim3(N / 32, M / 32), dim3(32), 0, stream>>>(
            A, lda, W, ldw, C, ldc, Cbf, ldcb, bias, Cadd, ldadd, cmod, K, act);
    };

    // 1) weight conversions fp32 -> bf16
    cvt(eWih_f, wih_f, (long)G * E);
    cvt(eWhh_f, whh_f, (long)G * HE);
    cvt(eWih_b, wih_b, (long)G * E);
    cvt(eWhh_b, whh_b, (long)G * HE);
    cvt(dWih,  dwih,  (long)G * (E + HD));
    cvt(dWhh,  dwhh,  (long)G * HD);
    cvt(W_h,   wh,    (long)HD * 2 * HE);
    cvt(W_c,   wc,    (long)HD * 2 * HE);
    cvt(W_att, watt,  (long)HD * 2 * HE);
    cvt(W_comb,wcomb, (long)HE * (2 * HE + HD));
    cvt(W_vocab,wvoc, (long)V * HD);

    // 2) embeddings
    build_src_k<<<2048, 256, 0, stream>>>(src, lens, src_emb, Xf, Xr);
    build_tgt_k<<<2048, 256, 0, stream>>>(tgt, tgt_emb, Ybuf);

    // 3) hoisted input-gate GEMMs: Gx = X @ Wih^T + b   (3072 x 4096, K=512)
    gemm2(Xf, E, wih_f, E, Gx_f, G, nullptr, 0, eb_f, nullptr, 0, 0, S * B, G, E, 0);
    gemm2(Xr, E, wih_b, E, Gx_b, G, nullptr, 0, eb_b, nullptr, 0, 0, S * B, G, E, 0);

    // 4) encoder recurrences (fwd then bwd)
    zero_bf16_k<<<64, 256, 0, stream>>>(hcat, (long)B * 2 * HE);
    for (int dir = 0; dir < 2; ++dir) {
        const bf16* whh = dir ? whh_b : whh_f;
        float* Gx = dir ? Gx_b : Gx_f;
        zero_f32_k<<<64, 256, 0, stream>>>(hbuf, (long)B * HE);
        zero_f32_k<<<64, 256, 0, stream>>>(cbuf, (long)B * HE);
        for (int t = 0; t < S; ++t) {
            gemm1(hcat + dir * HE, 2 * HE, whh, HE, gates, G, nullptr, 0,
                  nullptr, Gx + (size_t)t * B * G, G, 0, B, G, HE, 0);
            enc_pointwise_k<<<256, 256, 0, stream>>>(gates, hbuf, cbuf,
                hcat + dir * HE, ccat + dir * HE, enc_hid, lens, t, dir);
        }
    }

    // 5) decoder init
    gemm1(hcat, 2 * HE, wh, 2 * HE, nullptr, 0, dech0, HD, nullptr, nullptr, 0, 0,
          B, HD, 2 * HE, 0);
    gemm1(ccat, 2 * HE, wc, 2 * HE, decc0, HD, nullptr, 0, nullptr, nullptr, 0, 0,
          B, HD, 2 * HE, 0);
    gemm1(dech0, HD, dwhh, HD, Ghh0, G, nullptr, 0, db, nullptr, 0, 0, B, G, HD, 0);
    gemm2(enc_hid, 2 * HE, watt, 2 * HE, encproj, HD, nullptr, 0, nullptr,
          nullptr, 0, 0, B * S, HD, 2 * HE, 0);
    // Gy = Y @ dec_Wih[:, :E]^T + (dec_h0 @ dec_Whh^T + b)  [constant per step]
    gemm2(Ybuf, E, dwih, E + HD, Gy, G, nullptr, 0, nullptr, Ghh0, G, B,
          (T - 1) * B, G, E, 0);

    zero_bf16_k<<<64, 256, 0, stream>>>(zerob, (long)B * HD);

    // 6) decoder loop
    for (int t = 0; t < T - 1; ++t) {
        const bf16* oprev = t ? (comb + (size_t)(t - 1) * B * HD) : zerob;
        gemm1(oprev, HD, dwih + E, E + HD, gates, G, nullptr, 0,
              nullptr, Gy + (size_t)t * B * G, G, 0, B, G, HD, 0);
        dec_pointwise_k<<<256, 256, 0, stream>>>(gates, decc0, htbuf, Abuf);
        attention_k<<<B, 256, 0, stream>>>(htbuf, encproj, enc_hid, lens, Abuf);
        gemm1(Abuf, 2 * HE + HD, wcomb, 2 * HE + HD, nullptr, 0,
              comb + (size_t)t * B * HD, HD, nullptr, nullptr, 0, 0,
              B, HD, 2 * HE + HD, 1 /*tanh*/);
    }

    // 7) fused vocab projection + online log-softmax + gold gather
    zero_f32_k<<<1, 64, 0, stream>>>((float*)d_out, 64);
    vocab_loss_k<<<dim3((T - 1) * B / 16), dim3(256), 0, stream>>>(
        comb, wvoc, tgt, (float*)d_out);
}
